// IOHMM_43353399886214
// MI455X (gfx1250) — matrix operand order repaired and checked by reading, not compiled, not run on hardware
//
#include <hip/hip_runtime.h>
#include <hip/hip_bf16.h>

// IOHMM: B=16, T=512, K=32, V=512, D=64, H=128
#define Bsz 16
#define Tsz 512
#define Ksz 32
#define Vsz 512
#define Dsz 64
#define Hsz 128
#define ROWS (Bsz * Tsz)          // 8192 flattened (b,t) rows
#define KV   (Ksz * Vsz)          // 16384 emission cols
#define KK   (Ksz * Ksz)          // 1024 transition cols

typedef __attribute__((ext_vector_type(16))) _Float16 v16h;
typedef __attribute__((ext_vector_type(8)))  _Float16 v8h;
typedef __attribute__((ext_vector_type(8)))  float    v8f;

// ---------- WMMA fragment loaders (16x16x32 f16, contiguous k order, same for A and B) ----------
// A (16x32, MxK): lane row = lane%16.
//   lanes 0-15 : halves 0..15 -> K 0..15 contiguous
//   lanes 16-31: halves 0..15 -> K 16..31 contiguous
__device__ inline v16h load_a_frag(const _Float16* h, int row, int f, int lane) {
    const _Float16* base = h + row * Hsz + 32 * f;
    int off = (lane < 16) ? 0 : 16;
    v8h lo = *(const v8h*)(base + off);
    v8h hi = *(const v8h*)(base + off + 8);
    v16h a;
#pragma unroll
    for (int i = 0; i < 8; ++i) { a[i] = lo[i]; a[8 + i] = hi[i]; }
    return a;
}

// B (32x16, KxN) from a column-major (col*128 + k) fp16 weight copy:
//   lanes 0-15 : col = lane,    K = 0..15 contiguous
//   lanes 16-31: col = lane-16, K = 16..31 contiguous
__device__ inline v16h load_b_frag(const _Float16* wT, int col, int f, int lane) {
    const _Float16* p = wT + (size_t)col * Hsz + 32 * f + ((lane < 16) ? 0 : 16);
    return *(const v16h*)p;
}

// Butterfly logsumexp-combine of (m, s) pairs across each 16-lane half.
// Handles per-lane differing offsets m.
__device__ inline void lse_combine16(float& m, float& s) {
#pragma unroll
    for (int msk = 1; msk < 16; msk <<= 1) {
        float om = __shfl_xor(m, msk, 32);
        float os = __shfl_xor(s, msk, 32);
        float nm = fmaxf(m, om);
        s = s * __expf(m - nm) + os * __expf(om - nm);
        m = nm;
    }
}

// ---------- k0: fp32 -> fp16 column-major weight transposes ----------
__global__ void k0_prep(const float* __restrict__ W2, const float* __restrict__ E2,
                        _Float16* __restrict__ W2T, _Float16* __restrict__ E2T) {
    int idx = blockIdx.x * 256 + threadIdx.x;
    int stride = gridDim.x * 256;
    for (int i = idx; i < Hsz * KV; i += stride) {
        int h = i / KV, c = i % KV;
        E2T[(size_t)c * Hsz + h] = (_Float16)E2[i];
    }
    for (int i = idx; i < Hsz * KK; i += stride) {
        int h = i / KK, c = i % KK;
        W2T[(size_t)c * Hsz + h] = (_Float16)W2[i];
    }
}

// ---------- k1: embedding gather + both hidden MLPs (relu), fp16 out ----------
__global__ void k1_embed_mlp(const int* __restrict__ seq, const float* __restrict__ emb,
                             const float* __restrict__ W1, const float* __restrict__ b1,
                             const float* __restrict__ E1, const float* __restrict__ e1,
                             _Float16* __restrict__ h1, _Float16* __restrict__ h2) {
    int row = blockIdx.x;            // 0..8191 == b*T + t
    int j   = threadIdx.x;           // 0..127
    int b = row / Tsz, t = row % Tsz;
    int tok = (t == 0) ? 0 : seq[b * Tsz + t - 1];   // shifted input
    __shared__ float x[Dsz];
    if (j < Dsz) x[j] = emb[tok * Dsz + j];
    __syncthreads();
    float a1 = b1[j], a2 = e1[j];
#pragma unroll 8
    for (int d = 0; d < Dsz; ++d) {
        float xv = x[d];
        a1 += xv * W1[d * Hsz + j];
        a2 += xv * E1[d * Hsz + j];
    }
    h1[(size_t)row * Hsz + j] = (_Float16)fmaxf(a1, 0.f);
    h2[(size_t)row * Hsz + j] = (_Float16)fmaxf(a2, 0.f);
}

// ---------- k2: transition GEMM (WMMA) + fused softmax -> PROBABILITIES ----------
__global__ __launch_bounds__(256) void k2_trans(const _Float16* __restrict__ h1,
                                                const _Float16* __restrict__ W2T,
                                                const float* __restrict__ b2,
                                                float* __restrict__ trans_p) {
    int tile = blockIdx.x;           // 512 row-tiles of 16 rows
    int lane = threadIdx.x & 31;
    int wid  = threadIdx.x >> 5;     // 8 waves
    int row  = tile * 16 + (lane & 15);
    v16h A[4];
#pragma unroll
    for (int f = 0; f < 4; ++f) A[f] = load_a_frag(h1, row, f, lane);

    for (int kb = wid; kb < Ksz; kb += 8) {          // each wave: 4 source-states
        v8f acc0 = {}, acc1 = {};
        int c0 = kb * 32 + (lane & 15);
        int c1 = c0 + 16;
#pragma unroll
        for (int f = 0; f < 4; ++f) {
            v16h B0 = load_b_frag(W2T, c0, f, lane);
            acc0 = __builtin_amdgcn_wmma_f32_16x16x32_f16(false, A[f], false, B0,
                                                          (short)0, acc0, false, false);
        }
#pragma unroll
        for (int f = 0; f < 4; ++f) {
            v16h B1 = load_b_frag(W2T, c1, f, lane);
            acc1 = __builtin_amdgcn_wmma_f32_16x16x32_f16(false, A[f], false, B1,
                                                          (short)0, acc1, false, false);
        }
        float bias0 = b2[c0], bias1 = b2[c1];
        float z0[8], z1[8], m[8], s[8];
#pragma unroll
        for (int i = 0; i < 8; ++i) {
            z0[i] = acc0[i] + bias0;
            z1[i] = acc1[i] + bias1;
            float mi = fmaxf(z0[i], z1[i]);
            m[i] = mi;
            s[i] = __expf(z0[i] - mi) + __expf(z1[i] - mi);
        }
#pragma unroll
        for (int i = 0; i < 8; ++i) lse_combine16(m[i], s[i]);
#pragma unroll
        for (int i = 0; i < 8; ++i) {
            int r = (lane < 16) ? i : i + 8;          // C/D layout row
            float rcp = __builtin_amdgcn_rcpf(s[i]);
            size_t base = (size_t)(tile * 16 + r) * KK;
            trans_p[base + c0] = __expf(z0[i] - m[i]) * rcp;  // softmax prob
            trans_p[base + c1] = __expf(z1[i] - m[i]) * rcp;
        }
    }
}

// ---------- k3: emission GEMM (WMMA), fused single-exp logsumexp over V,
//             gather observed-token prob -> emit_p = exp(log_softmax[tgt]) ----------
__global__ __launch_bounds__(256) void k3_emit(const _Float16* __restrict__ h2,
                                               const _Float16* __restrict__ E2T,
                                               const float* __restrict__ e2,
                                               const int* __restrict__ seq,
                                               float* __restrict__ emit_p) {
    int tile = blockIdx.x;           // 512 row-tiles
    int lane = threadIdx.x & 31;
    int wid  = threadIdx.x >> 5;
    int row  = tile * 16 + (lane & 15);
    v16h A[4];
#pragma unroll
    for (int f = 0; f < 4; ++f) A[f] = load_a_frag(h2, row, f, lane);

    int tgt[8];
#pragma unroll
    for (int i = 0; i < 8; ++i) {
        int r = (lane < 16) ? i : i + 8;
        tgt[i] = seq[tile * 16 + r];                  // observed token per row
    }

    for (int k = wid; k < Ksz; k += 8) {              // each wave: 4 states
        float off[8], s[8], zt[8];
        // ---- peel vt = 0: establish per-row offset (no running max needed) ----
        {
            int v   = lane & 15;
            int col = k * Vsz + v;
            v8f acc = {};
#pragma unroll
            for (int f = 0; f < 4; ++f) {
                v16h Bf = load_b_frag(E2T, col, f, lane);
                acc = __builtin_amdgcn_wmma_f32_16x16x32_f16(false, A[f], false, Bf,
                                                             (short)0, acc, false, false);
            }
            float bias = e2[col];
#pragma unroll
            for (int i = 0; i < 8; ++i) {
                float z = acc[i] + bias;
                off[i] = z;
                s[i]   = 1.0f;
                zt[i]  = (v == tgt[i]) ? z : -__builtin_inff();
            }
        }
        // ---- remaining 31 tiles: one exp per element ----
        for (int vt = 1; vt < Vsz / 16; ++vt) {
            int v   = vt * 16 + (lane & 15);
            int col = k * Vsz + v;
            v8f acc = {};
#pragma unroll
            for (int f = 0; f < 4; ++f) {
                v16h Bf = load_b_frag(E2T, col, f, lane);
                acc = __builtin_amdgcn_wmma_f32_16x16x32_f16(false, A[f], false, Bf,
                                                             (short)0, acc, false, false);
            }
            float bias = e2[col];
#pragma unroll
            for (int i = 0; i < 8; ++i) {
                float z = acc[i] + bias;
                if (v == tgt[i]) zt[i] = z;
                s[i] += __expf(z - off[i]);
            }
        }
#pragma unroll
        for (int i = 0; i < 8; ++i) {
            lse_combine16(off[i], s[i]);
#pragma unroll
            for (int msk = 1; msk < 16; msk <<= 1)
                zt[i] = fmaxf(zt[i], __shfl_xor(zt[i], msk, 32));
        }
        if ((lane & 15) == 0) {
#pragma unroll
            for (int i = 0; i < 8; ++i) {
                int r = (lane < 16) ? i : i + 8;
                // exp(zt - lse) = exp(zt - off)/s : observed-token emission prob
                emit_p[(size_t)(tile * 16 + r) * Ksz + k] =
                    __expf(zt[i] - off[i]) * __builtin_amdgcn_rcpf(s[i]);
            }
        }
    }
}

// ---------- k4: scaled linear-space forward recursion (no trans ops in loop) ----
// alpha_t tracked as p (Σp in [0.5,1)) plus power-of-two exponent eacc.
__global__ void k4_forward(const float* __restrict__ trans_p,
                           const float* __restrict__ emit_p,
                           const float* __restrict__ init_logits,
                           const int* __restrict__ lengths,
                           float* __restrict__ out) {
    int b = blockIdx.x;              // 16 batches
    int j = threadIdx.x;             // 32 states (one wave)
    // softmax(init_logits)
    float il = init_logits[j];
    float m = il;
#pragma unroll
    for (int msk = 1; msk < 32; msk <<= 1) m = fmaxf(m, __shfl_xor(m, msk, 32));
    float w = __expf(il - m);
    float ws = w;
#pragma unroll
    for (int msk = 1; msk < 32; msk <<= 1) ws += __shfl_xor(ws, msk, 32);
    float ip = w * __builtin_amdgcn_rcpf(ws);

    // alpha0 (linear): q = softmax(init) * emit_p[t=0]
    float q = ip * emit_p[(size_t)(b * Tsz) * Ksz + j];
    float sc = q;
#pragma unroll
    for (int msk = 1; msk < 32; msk <<= 1) sc += __shfl_xor(sc, msk, 32);
    int e;
    frexpf(sc, &e);                  // sc = mant * 2^e, mant in [0.5,1)
    float p   = ldexpf(q, -e);
    int  eacc = e;
    int  len  = lengths[b];
    __shared__ float sA[Ksz];

    for (int t = 1; t < Tsz; ++t) {
        sA[j] = p;
        __syncthreads();
        const float* Pt = trans_p + (size_t)(b * Tsz + t) * KK;
        float a0 = 0.f, a1 = 0.f, a2 = 0.f, a3 = 0.f;   // 4-way ILP matvec
#pragma unroll
        for (int i = 0; i < Ksz; i += 4) {
            a0 = fmaf(sA[i + 0], Pt[(i + 0) * Ksz + j], a0);
            a1 = fmaf(sA[i + 1], Pt[(i + 1) * Ksz + j], a1);
            a2 = fmaf(sA[i + 2], Pt[(i + 2) * Ksz + j], a2);
            a3 = fmaf(sA[i + 3], Pt[(i + 3) * Ksz + j], a3);
        }
        float q2 = ((a0 + a1) + (a2 + a3)) * emit_p[(size_t)(b * Tsz + t) * Ksz + j];
        float sc2 = q2;
#pragma unroll
        for (int msk = 1; msk < 32; msk <<= 1) sc2 += __shfl_xor(sc2, msk, 32);
        int e2;
        frexpf(sc2, &e2);                         // v_frexp_exp (VALU, not trans)
        bool valid = (t < len);
        p    = valid ? ldexpf(q2, -e2) : p;       // v_ldexp (VALU)
        eacc = valid ? eacc + e2 : eacc;
        __syncthreads();
    }
    float sp = p;
#pragma unroll
    for (int msk = 1; msk < 32; msk <<= 1) sp += __shfl_xor(sp, msk, 32);
    if (j == 0)
        out[b] = -((float)eacc * 0.6931471805599453f + __logf(sp));
}

extern "C" void kernel_launch(void* const* d_in, const int* in_sizes, int n_in,
                              void* d_out, int out_size, void* d_ws, size_t ws_size,
                              hipStream_t stream) {
    const int*   seq         = (const int*)d_in[0];
    const int*   lengths     = (const int*)d_in[1];
    const float* emb         = (const float*)d_in[2];
    const float* W1          = (const float*)d_in[3];
    const float* b1          = (const float*)d_in[4];
    const float* W2          = (const float*)d_in[5];
    const float* b2          = (const float*)d_in[6];
    const float* E1          = (const float*)d_in[7];
    const float* e1          = (const float*)d_in[8];
    const float* E2          = (const float*)d_in[9];
    const float* e2          = (const float*)d_in[10];
    const float* init_logits = (const float*)d_in[11];
    float* out = (float*)d_out;

    char* ws = (char*)d_ws;
    size_t off = 0;
    auto alloc = [&](size_t bytes) -> void* {
        void* p = ws + off;
        off += (bytes + 255) & ~(size_t)255;
        return p;
    };
    _Float16* E2T     = (_Float16*)alloc((size_t)KV * Hsz * 2);      // 4 MB
    _Float16* W2T     = (_Float16*)alloc((size_t)KK * Hsz * 2);      // 256 KB
    _Float16* h1      = (_Float16*)alloc((size_t)ROWS * Hsz * 2);    // 2 MB
    _Float16* h2      = (_Float16*)alloc((size_t)ROWS * Hsz * 2);    // 2 MB
    float*    trans_p = (float*)alloc((size_t)ROWS * KK * 4);        // 32 MB
    float*    emit_p  = (float*)alloc((size_t)ROWS * Ksz * 4);       // 1 MB

    k0_prep<<<512, 256, 0, stream>>>(W2, E2, W2T, E2T);
    k1_embed_mlp<<<ROWS, Hsz, 0, stream>>>(seq, emb, W1, b1, E1, e1, h1, h2);
    k2_trans<<<ROWS / 16, 256, 0, stream>>>(h1, W2T, b2, trans_p);
    k3_emit<<<ROWS / 16, 256, 0, stream>>>(h2, E2T, e2, seq, emit_p);
    k4_forward<<<Bsz, Ksz, 0, stream>>>(trans_p, emit_p, init_logits, lengths, out);
}
